// PremiseSelectionModel_62646392979490
// MI455X (gfx1250) — matrix operand order repaired and checked by reading, not compiled, not run on hardware
//
#include <hip/hip_runtime.h>
#include <hip/hip_bf16.h>
#include <math.h>

// ---------------------------------------------------------------------------
// Premise-selection DAG-GNN for MI455X (gfx1250, wave32).
// GEMMs run on v_wmma_f32_16x16x32_bf16 (f32 -> bf16 conversion on the fly;
// bf16 keeps full f32 exponent range). Gate GEMM fuses the edge gather
// (concat [h[row]|e|h[col]] built per 32-wide K chunk in LDS) and the
// e@W1 -> sigmoid gate epilogue (wave shfl reductions).
// f32x2 -> packed bf16 uses v_add (round) + v_perm_b32 (pack): 3 VALU ops.
// Next-K W tiles are prefetched with global_prefetch_b8.
// ---------------------------------------------------------------------------

typedef __attribute__((ext_vector_type(16))) __bf16 v16bf;
typedef __attribute__((ext_vector_type(8)))  float  v8f;

union FragAB { v16bf v; unsigned int u[8]; };

__device__ __forceinline__ unsigned int pack_bf16x2(float a, float b) {
  // round-to-nearest (+0x8000) then byte-select both high halves in one
  // v_perm_b32: result = { hi16(b), hi16(a) }.
  unsigned int ua = __float_as_uint(a) + 0x8000u;
  unsigned int ub = __float_as_uint(b) + 0x8000u;
  return __builtin_amdgcn_perm(ub, ua, 0x07060302u);
}

// ---------------------------------------------------------------------------
// 1) argmax one-hot -> embedding row
// ---------------------------------------------------------------------------
__global__ __launch_bounds__(256) void embed_nodes_kernel(
    const float* __restrict__ xoh, const float* __restrict__ emb,
    float* __restrict__ h, int Nn, int V) {
  int nId = blockIdx.x * 8 + (threadIdx.x >> 5);
  if (nId >= Nn) return;
  int lane = threadIdx.x & 31;
  const float* row = xoh + (size_t)nId * V;
  float best = -3.4e38f; int bidx = 0x7FFFFFFF;
  for (int c = lane; c < V; c += 32) {
    float v = row[c];
    if (v > best) { best = v; bidx = c; }
  }
  for (int s = 16; s >= 1; s >>= 1) {
    float ov = __shfl_xor(best, s, 32);
    int   oi = __shfl_xor(bidx, s, 32);
    if (ov > best || (ov == best && oi < bidx)) { best = ov; bidx = oi; }
  }
  const float* er = emb + (size_t)bidx * 128;
  float* hr = h + (size_t)nId * 128;
  #pragma unroll
  for (int i = 0; i < 4; ++i) hr[lane + 32 * i] = er[lane + 32 * i];
}

__global__ __launch_bounds__(256) void embed_edges_kernel(
    const float* __restrict__ eoh, const float* __restrict__ emb,
    float* __restrict__ e, int Etot) {
  int id = blockIdx.x * blockDim.x + threadIdx.x;
  if (id >= Etot) return;
  const float* row = eoh + (size_t)id * 8;
  float best = row[0]; int bidx = 0;
  #pragma unroll
  for (int c = 1; c < 8; ++c) { float v = row[c]; if (v > best) { best = v; bidx = c; } }
  const float4* er = (const float4*)(emb + (size_t)bidx * 64);
  float4* out = (float4*)(e + (size_t)id * 64);
  #pragma unroll
  for (int i = 0; i < 16; ++i) out[i] = er[i];
}

// ---------------------------------------------------------------------------
// 2) Gate GEMM: e_next = [h[i0] | e | h[i1]] @ W (E x 320 @ 320 x 64), fused
//    gate = sigmoid(e_next @ W1). One block = 128 edge rows; 8 waves, each
//    wave owns a 16-row M-subtile and the full N=64 (4 accumulators).
// ---------------------------------------------------------------------------
__global__ __launch_bounds__(256) void gate_gemm_kernel(
    const float* __restrict__ h, const float* __restrict__ e_cur,
    const int* __restrict__ erow, const int* __restrict__ ecol,
    const float* __restrict__ W_in,  const float* __restrict__ W1_in,
    const float* __restrict__ W_out, const float* __restrict__ W1_out,
    float* __restrict__ e_next, float* __restrict__ wgate, int E) {
  const int dir = blockIdx.y;
  const int m0  = blockIdx.x * 128;
  const int t = threadIdx.x, lane = t & 31, wv = t >> 5;

  const float* W   = dir ? W_out  : W_in;
  const float* W1  = dir ? W1_out : W1_in;
  const int*  idx0 = dir ? ecol : erow;
  const int*  idx1 = dir ? erow : ecol;
  const float* ebase = e_cur + (size_t)dir * (size_t)E * 64;
  float* enext = e_next + (size_t)dir * (size_t)E * 64;
  float* wg    = wgate  + (size_t)dir * E;

  __shared__ unsigned int ldsA[128 * 16];   // 128 rows x 32 bf16 (packed)
  __shared__ unsigned int ldsBt[64 * 16];   // 64 cols  x 32 bf16 along K
  __shared__ float ldsW1[64];
  if (t < 64) ldsW1[t] = W1[t];

  v8f acc[4];
  #pragma unroll
  for (int j = 0; j < 4; ++j)
    #pragma unroll
    for (int r = 0; r < 8; ++r) acc[j][r] = 0.0f;

  for (int ks = 0; ks < 10; ++ks) {
    const int kk = ks * 32;
    { // stage A (gathered concat segment)
      const int rloc = t >> 1, half = t & 1;
      const int r = m0 + rloc;
      const int er = (r < E) ? r : (E - 1);
      const float* src;
      if (kk < 128)      src = h + (size_t)idx0[er] * 128 + kk;
      else if (kk < 192) src = ebase + (size_t)er * 64 + (kk - 128);
      else               src = h + (size_t)idx1[er] * 128 + (kk - 192);
      src += half * 16;
      #pragma unroll
      for (int i = 0; i < 8; ++i)
        ldsA[rloc * 16 + half * 8 + i] = pack_bf16x2(src[2 * i], src[2 * i + 1]);
    }
    { // stage Bt: W rows kk..kk+31, transposed to [col][k]
      const int c = t & 63, kb = (t >> 6) * 8;
      #pragma unroll
      for (int i = 0; i < 4; ++i) {
        float b0 = W[(size_t)(kk + kb + 2 * i)     * 64 + c];
        float b1 = W[(size_t)(kk + kb + 2 * i + 1) * 64 + c];
        ldsBt[c * 16 + (kb >> 1) + i] = pack_bf16x2(b0, b1);
      }
      if (kk + 32 < 320)  // prefetch next W k-chunk (global_prefetch_b8)
        __builtin_prefetch(&W[(size_t)(kk + 32) * 64 + (size_t)t * 8], 0, 0);
    }
    __syncthreads();
    FragAB fa;
    { // A frag: lane<16 -> K {0..7,16..23}; lane>=16 -> K {8..15,24..31}
      const int m = wv * 16 + (lane & 15);
      const int kd = (lane < 16) ? 0 : 4;
      #pragma unroll
      for (int i = 0; i < 4; ++i) fa.u[i]     = ldsA[m * 16 + kd + i];
      #pragma unroll
      for (int i = 0; i < 4; ++i) fa.u[4 + i] = ldsA[m * 16 + 8 + kd + i];
    }
    #pragma unroll
    for (int j = 0; j < 4; ++j) {
      FragAB fb;   // B frag: lane<16 -> K {0..7,16..23}; lane>=16 -> K {8..15,24..31}
      const int n = 16 * j + (lane & 15);
      const int kd = (lane < 16) ? 0 : 4;
      #pragma unroll
      for (int i = 0; i < 8; ++i) fb.u[i] = ldsBt[n * 16 + kd + (i & 3) + ((i >> 2) << 3)];
      acc[j] = __builtin_amdgcn_wmma_f32_16x16x32_bf16(
          false, fa.v, false, fb.v, (short)0, acc[j], false, false);
    }
    __syncthreads();
  }

  // epilogue: store e_next tile, fused gate = sigmoid(row . W1)
  const int mrow = (lane < 16) ? 0 : 8;
  const int ncol = lane & 15;
  float pr[8];
  #pragma unroll
  for (int r = 0; r < 8; ++r) pr[r] = 0.0f;
  #pragma unroll
  for (int j = 0; j < 4; ++j) {
    const int gcol = 16 * j + ncol;
    const float w1v = ldsW1[gcol];
    #pragma unroll
    for (int r = 0; r < 8; ++r) {
      const int gr = m0 + wv * 16 + mrow + r;
      const float val = acc[j][r];
      if (gr < E) enext[(size_t)gr * 64 + gcol] = val;
      pr[r] += val * w1v;
    }
  }
  #pragma unroll
  for (int s = 1; s < 16; s <<= 1)
    #pragma unroll
    for (int r = 0; r < 8; ++r) pr[r] += __shfl_xor(pr[r], s, 32);
  if (ncol == 0) {
    #pragma unroll
    for (int r = 0; r < 8; ++r) {
      const int gr = m0 + wv * 16 + mrow + r;
      if (gr < E) wg[gr] = 1.0f / (1.0f + __expf(-pr[r]));
    }
  }
}

// ---------------------------------------------------------------------------
// 3) Node GEMM: xi = x @ Wg (Nn x 128 @ 128 x 128). Same tiling; each wave
//    owns 16 rows x full 128 cols (8 accumulators), K = 4 steps of 32.
// ---------------------------------------------------------------------------
__global__ __launch_bounds__(256) void node_gemm_kernel(
    const float* __restrict__ x,
    const float* __restrict__ Wg_in, const float* __restrict__ Wg_out,
    float* __restrict__ xi, float* __restrict__ xo, int Nn) {
  const int dir = blockIdx.y;
  const float* Wg = dir ? Wg_out : Wg_in;
  float* out = dir ? xo : xi;
  const int m0 = blockIdx.x * 128;
  const int t = threadIdx.x, lane = t & 31, wv = t >> 5;

  __shared__ unsigned int ldsA[128 * 16];
  __shared__ unsigned int ldsBt[128 * 16];

  v8f acc[8];
  #pragma unroll
  for (int j = 0; j < 8; ++j)
    #pragma unroll
    for (int r = 0; r < 8; ++r) acc[j][r] = 0.0f;

  for (int ks = 0; ks < 4; ++ks) {
    const int kk = ks * 32;
    {
      const int rloc = t >> 1, half = t & 1;
      const int r = m0 + rloc;
      const int cr = (r < Nn) ? r : (Nn - 1);
      const float* src = x + (size_t)cr * 128 + kk + half * 16;
      #pragma unroll
      for (int i = 0; i < 8; ++i)
        ldsA[rloc * 16 + half * 8 + i] = pack_bf16x2(src[2 * i], src[2 * i + 1]);
    }
    {
      const int c = t & 127, kb = (t >> 7) * 16;  // kb = 0 or 16
      #pragma unroll
      for (int i = 0; i < 8; ++i) {
        float b0 = Wg[(size_t)(kk + kb + 2 * i)     * 128 + c];
        float b1 = Wg[(size_t)(kk + kb + 2 * i + 1) * 128 + c];
        ldsBt[c * 16 + (kb >> 1) + i] = pack_bf16x2(b0, b1);
      }
      if (kk + 32 < 128)  // prefetch next Wg k-chunk
        __builtin_prefetch(&Wg[(size_t)(kk + 32) * 128 + (size_t)t * 16], 0, 0);
    }
    __syncthreads();
    FragAB fa;
    {
      const int m = wv * 16 + (lane & 15);
      const int kd = (lane < 16) ? 0 : 4;
      #pragma unroll
      for (int i = 0; i < 4; ++i) fa.u[i]     = ldsA[m * 16 + kd + i];
      #pragma unroll
      for (int i = 0; i < 4; ++i) fa.u[4 + i] = ldsA[m * 16 + 8 + kd + i];
    }
    #pragma unroll
    for (int j = 0; j < 8; ++j) {
      FragAB fb;
      const int n = 16 * j + (lane & 15);
      const int kd = (lane < 16) ? 0 : 4;
      #pragma unroll
      for (int i = 0; i < 8; ++i) fb.u[i] = ldsBt[n * 16 + kd + (i & 3) + ((i >> 2) << 3)];
      acc[j] = __builtin_amdgcn_wmma_f32_16x16x32_bf16(
          false, fa.v, false, fb.v, (short)0, acc[j], false, false);
    }
    __syncthreads();
  }
  const int mrow = (lane < 16) ? 0 : 8;
  const int ncol = lane & 15;
  #pragma unroll
  for (int j = 0; j < 8; ++j)
    #pragma unroll
    for (int r = 0; r < 8; ++r) {
      const int gr = m0 + wv * 16 + mrow + r;
      if (gr < Nn) out[(size_t)gr * 128 + 16 * j + ncol] = acc[j][r];
    }
}

// ---------------------------------------------------------------------------
// 4) x_new = x + bg_in + bg_out, then scatter-add gated messages.
// ---------------------------------------------------------------------------
__global__ void node_init_kernel(const float* __restrict__ x,
                                 const float* __restrict__ bg_in,
                                 const float* __restrict__ bg_out,
                                 float* __restrict__ xnew, int Nn) {
  int i = blockIdx.x * blockDim.x + threadIdx.x;
  if (i < Nn * 128) {
    int c = i & 127;
    xnew[i] = x[i] + bg_in[c] + bg_out[c];
  }
}

__global__ __launch_bounds__(256) void scatter_kernel(
    const int* __restrict__ erow, const int* __restrict__ ecol,
    const float* __restrict__ xi, const float* __restrict__ xo,
    const float* __restrict__ wgate, float* __restrict__ xnew, int E) {
  const int dir = blockIdx.y;
  const int e = blockIdx.x * 8 + (threadIdx.x >> 5);
  if (e >= E) return;
  const int lane = threadIdx.x & 31;
  const int src = dir ? ecol[e] : erow[e];
  const int dst = dir ? erow[e] : ecol[e];
  const float* xs = dir ? xo : xi;
  const float wv = wgate[(size_t)dir * E + e];
  const float* srcp = xs + (size_t)src * 128;
  float* dstp = xnew + (size_t)dst * 128;
  #pragma unroll
  for (int i = 0; i < 4; ++i) {
    const int c = lane + 32 * i;
    atomicAdd(&dstp[c], wv * srcp[c]);
  }
}

// ---------------------------------------------------------------------------
// 5) Mean pool (sum + count via atomics; divide in classifier).
// ---------------------------------------------------------------------------
__global__ __launch_bounds__(256) void pool_kernel(
    const float* __restrict__ x, const int* __restrict__ batch,
    float* __restrict__ g, float* __restrict__ cnt, int Nn) {
  const int nId = blockIdx.x * 8 + (threadIdx.x >> 5);
  if (nId >= Nn) return;
  const int lane = threadIdx.x & 31;
  const int b = batch[nId];
  const float* xp = x + (size_t)nId * 128;
  float* gp = g + (size_t)b * 128;
  #pragma unroll
  for (int i = 0; i < 4; ++i) {
    const int c = lane + 32 * i;
    atomicAdd(&gp[c], xp[c]);
  }
  if (lane == 0) atomicAdd(&cnt[b], 1.0f);
}

// ---------------------------------------------------------------------------
// 6) Classifier + log_softmax + mean NLL. One block; one wave per graph.
// ---------------------------------------------------------------------------
__global__ __launch_bounds__(256) void classifier_kernel(
    const float* __restrict__ g_s, const float* __restrict__ cnt_s,
    const float* __restrict__ g_t, const float* __restrict__ cnt_t,
    const float* __restrict__ W0, const float* __restrict__ b0,
    const float* __restrict__ W1, const float* __restrict__ b1,
    const int* __restrict__ y, float* __restrict__ out, int G) {
  __shared__ float perGraph[128];
  const int lane = threadIdx.x & 31, wv = threadIdx.x >> 5;
  for (int gi = wv; gi < G; gi += 8) {
    const float inv_s = 1.0f / fmaxf(cnt_s[gi], 1.0f);
    const float inv_t = 1.0f / fmaxf(cnt_t[gi], 1.0f);
    float h1[4];
    #pragma unroll
    for (int q = 0; q < 4; ++q) {
      const int o = lane * 4 + q;
      float a = b0[o];
      for (int c = 0; c < 128; ++c)
        a += g_s[gi * 128 + c] * inv_s * W0[c * 128 + o];
      for (int c = 0; c < 128; ++c)
        a += g_t[gi * 128 + c] * inv_t * W0[(128 + c) * 128 + o];
      h1[q] = fmaxf(a, 0.0f);
    }
    float l0 = 0.0f, l1 = 0.0f;
    #pragma unroll
    for (int q = 0; q < 4; ++q) {
      const int o = lane * 4 + q;
      l0 += h1[q] * W1[o * 2 + 0];
      l1 += h1[q] * W1[o * 2 + 1];
    }
    for (int s = 16; s >= 1; s >>= 1) {
      l0 += __shfl_xor(l0, s, 32);
      l1 += __shfl_xor(l1, s, 32);
    }
    if (lane == 0) {
      l0 += b1[0]; l1 += b1[1];
      const float m = fmaxf(l0, l1);
      const float lse = m + __logf(__expf(l0 - m) + __expf(l1 - m));
      const float lp = ((y[gi] == 0) ? l0 : l1) - lse;
      perGraph[gi] = -lp;
    }
  }
  __syncthreads();
  if (threadIdx.x == 0) {
    float s = 0.0f;
    for (int gi = 0; gi < G; ++gi) s += perGraph[gi];
    out[0] = s / (float)G;
  }
}

// ---------------------------------------------------------------------------
// Host orchestration
// ---------------------------------------------------------------------------
extern "C" void kernel_launch(void* const* d_in, const int* in_sizes, int n_in,
                              void* d_out, int out_size, void* d_ws, size_t ws_size,
                              hipStream_t stream) {
  const int Nn = 100000, E = 300000, G = 128;
  const float* x_s      = (const float*)d_in[0];
  const float* x_t      = (const float*)d_in[1];
  const float* ea_s     = (const float*)d_in[2];
  const float* ea_t     = (const float*)d_in[3];
  const int*   ei_s     = (const int*)d_in[4];
  const int*   ei_t     = (const int*)d_in[5];
  const int*   b_s      = (const int*)d_in[6];
  const int*   b_t      = (const int*)d_in[7];
  const int*   y        = (const int*)d_in[8];
  const float* node_emb = (const float*)d_in[9];
  const float* edge_emb = (const float*)d_in[10];
  const float* W_in_0   = (const float*)d_in[11];
  const float* W_in_1   = (const float*)d_in[12];
  const float* W_out_0  = (const float*)d_in[13];
  const float* W_out_1  = (const float*)d_in[14];
  const float* Wg_in    = (const float*)d_in[15];
  const float* bg_in    = (const float*)d_in[16];
  const float* Wg_out   = (const float*)d_in[17];
  const float* bg_out   = (const float*)d_in[18];
  const float* cls_W0   = (const float*)d_in[19];
  const float* cls_b0   = (const float*)d_in[20];
  const float* cls_W1   = (const float*)d_in[21];
  const float* cls_b1   = (const float*)d_in[22];

  float* ws = (float*)d_ws;
  size_t o = 0;
  float* hA = ws + o; o += (size_t)Nn * 128;
  float* hB = ws + o; o += (size_t)Nn * 128;
  float* eA = ws + o; o += (size_t)2 * E * 64;
  float* eB = ws + o; o += (size_t)2 * E * 64;
  float* wg = ws + o; o += (size_t)2 * E;
  float* xi = ws + o; o += (size_t)Nn * 128;
  float* xo = ws + o; o += (size_t)Nn * 128;
  float* gpool = ws + o;
  float* g_s = gpool;
  float* g_t = gpool + (size_t)G * 128;
  float* cnt_s = gpool + (size_t)2 * G * 128;
  float* cnt_t = cnt_s + G;

  hipMemsetAsync(gpool, 0, ((size_t)2 * G * 128 + 2 * G) * sizeof(float), stream);

  const int etiles = (E + 127) / 128;
  const int ntiles = (Nn + 127) / 128;

  for (int side = 0; side < 2; ++side) {
    const float* xoh  = side ? x_t  : x_s;
    const float* eoh  = side ? ea_t : ea_s;
    const int* erow   = side ? ei_t : ei_s;   // edge_index[0]
    const int* ecol   = erow + E;             // edge_index[1]
    const int* batch  = side ? b_t : b_s;
    float* gp = side ? g_t : g_s;
    float* cp = side ? cnt_t : cnt_s;

    float* hcur = hA; float* hnext = hB;
    float* ecur = eA; float* enext = eB;

    embed_nodes_kernel<<<(Nn + 7) / 8, 256, 0, stream>>>(xoh, node_emb, hcur, Nn, 200);
    embed_edges_kernel<<<(2 * E + 255) / 256, 256, 0, stream>>>(eoh, edge_emb, ecur, 2 * E);

    for (int k = 0; k < 3; ++k) {
      gate_gemm_kernel<<<dim3(etiles, 2), 256, 0, stream>>>(
          hcur, ecur, erow, ecol,
          W_in_0  + (size_t)k * 320 * 64, W_in_1  + (size_t)k * 64,
          W_out_0 + (size_t)k * 320 * 64, W_out_1 + (size_t)k * 64,
          enext, wg, E);
      node_gemm_kernel<<<dim3(ntiles, 2), 256, 0, stream>>>(
          hcur, Wg_in + (size_t)k * 128 * 128, Wg_out + (size_t)k * 128 * 128,
          xi, xo, Nn);
      node_init_kernel<<<(Nn * 128 + 255) / 256, 256, 0, stream>>>(
          hcur, bg_in + (size_t)k * 128, bg_out + (size_t)k * 128, hnext, Nn);
      scatter_kernel<<<dim3((E + 7) / 8, 2), 256, 0, stream>>>(
          erow, ecol, xi, xo, wg, hnext, E);
      float* tmp = hcur; hcur = hnext; hnext = tmp;
      tmp = ecur; ecur = enext; enext = tmp;
    }
    pool_kernel<<<(Nn + 7) / 8, 256, 0, stream>>>(hcur, batch, gp, cp, Nn);
  }

  classifier_kernel<<<1, 256, 0, stream>>>(
      g_s, cnt_s, g_t, cnt_t, cls_W0, cls_b0, cls_W1, cls_b1,
      y, (float*)d_out, G);
}
